// XNOR_MLP_9869834846709
// MI455X (gfx1250) — compile-verified, matrix operand order checked by reading, not run on hardware
//
#include <hip/hip_runtime.h>
#include <hip/hip_bf16.h>

typedef __bf16 bf16_t;
typedef __attribute__((ext_vector_type(16))) __bf16 v16bf;
typedef __attribute__((ext_vector_type(8)))  float  v8f;

#define BN_EPS 1e-5f
#define Bsz 65536
#define KIN 784
#define KP  800      // K padded to multiple of 32 (and of 8: full 16B chunks)
#define HID 1024
#define NCLS 10

// -------- CDNA5 async global->LDS copy (16B per lane), ASYNCcnt-tracked ----
__device__ __forceinline__ void async_copy_b128(void* lds_ptr, const void* gptr) {
    const unsigned lds_off = (unsigned)(unsigned long long)lds_ptr; // LDS offset in low 32b
    const unsigned long long ga = (unsigned long long)gptr;
    asm volatile("global_load_async_to_lds_b128 %0, %1, off"
                 :: "v"(lds_off), "v"(ga) : "memory");
}
__device__ __forceinline__ void wait_async0() {
    asm volatile("s_wait_asynccnt 0x0" ::: "memory");
}

// ---------------- init: zero BN accumulators (must re-zero every call) -----
__global__ __launch_bounds__(256) void k_init_stats(float* sum, float* sumsq) {
    for (int i = threadIdx.x; i < HID; i += 256) { sum[i] = 0.f; sumsq[i] = 0.f; }
}

// ---------------- convert x: f32 -> bf16, zero-padded to KP ----------------
__global__ __launch_bounds__(256)
void k_convert_x(const float* __restrict__ x, bf16_t* __restrict__ xb) {
    const int idx = blockIdx.x * 256 + threadIdx.x;
    const int row = idx / (KP / 8);
    const int kc  = (idx % (KP / 8)) * 8;
    union { bf16_t h[8]; int4 v; } u;
    if (kc < KIN) {
        const float4 a = *(const float4*)&x[(size_t)row * KIN + kc];
        const float4 b = *(const float4*)&x[(size_t)row * KIN + kc + 4];
        u.h[0] = (bf16_t)a.x; u.h[1] = (bf16_t)a.y;
        u.h[2] = (bf16_t)a.z; u.h[3] = (bf16_t)a.w;
        u.h[4] = (bf16_t)b.x; u.h[5] = (bf16_t)b.y;
        u.h[6] = (bf16_t)b.z; u.h[7] = (bf16_t)b.w;
    } else {
        u.v = make_int4(0, 0, 0, 0);
    }
    *(int4*)&xb[(size_t)row * KP + kc] = u.v;
}

// ---------------- prep w1: alpha1[row] = mean|w1|, wb1 = sign(w1) as bf16 --
__global__ __launch_bounds__(256) void k_prep_w1(const float* __restrict__ w1,
                                                 bf16_t* __restrict__ wb1,
                                                 float* __restrict__ alpha1) {
    const int row = blockIdx.x;
    __shared__ float red[256];
    float s = 0.f;
    for (int k = threadIdx.x; k < KIN; k += 256) s += fabsf(w1[(size_t)row * KIN + k]);
    red[threadIdx.x] = s; __syncthreads();
    for (int o = 128; o > 0; o >>= 1) {
        if (threadIdx.x < o) red[threadIdx.x] += red[threadIdx.x + o];
        __syncthreads();
    }
    if (threadIdx.x == 0) alpha1[row] = red[0] / (float)KIN;
    for (int k = threadIdx.x; k < KP; k += 256) {
        float sg = 0.f;
        if (k < KIN) {
            float w = w1[(size_t)row * KIN + k];
            sg = (w > 0.f) ? 1.f : ((w < 0.f) ? -1.f : 0.f);
        }
        wb1[(size_t)row * KP + k] = (bf16_t)sg;
    }
}

// ---------------- prep w2: alpha2[c] = mean|w2|, bitmask of sign(w2) -------
__global__ __launch_bounds__(256) void k_prep_w2(const float* __restrict__ w2,
                                                 unsigned* __restrict__ w2bits,
                                                 float* __restrict__ alpha2) {
    const int row = blockIdx.x;           // class 0..9
    __shared__ float red[256];
    float s = 0.f;
    for (int k = threadIdx.x; k < HID; k += 256) s += fabsf(w2[(size_t)row * HID + k]);
    red[threadIdx.x] = s; __syncthreads();
    for (int o = 128; o > 0; o >>= 1) {
        if (threadIdx.x < o) red[threadIdx.x] += red[threadIdx.x + o];
        __syncthreads();
    }
    if (threadIdx.x == 0) alpha2[row] = red[0] / (float)HID;
    if (threadIdx.x < 32) {
        unsigned word = 0u;
        for (int i = 0; i < 32; ++i) {
            float w = w2[(size_t)row * HID + threadIdx.x * 32 + i];
            if (w >= 0.f) word |= (1u << i);
        }
        w2bits[row * 32 + threadIdx.x] = word;
    }
}

// ---------------- GEMM1: h = (xb bf16) @ wb1^T * alpha1  via WMMA ----------
// block tile 128(M) x 128(N), 8 waves as 2x4, wave tile 64x32 = 4x2 WMMAs.
// Double-buffered LDS filled by CDNA5 async global->LDS DMA copies.
#define TM 128
#define TN 128
#define TK 32
#define LDA (TK + 8)   // bf16 elements; row stride 80B (16B-aligned, bank-skewed)

__global__ __launch_bounds__(256)
void k_gemm1_wmma(const bf16_t* __restrict__ xb, const bf16_t* __restrict__ wb1,
                  const float* __restrict__ alpha1, float* __restrict__ h) {
    __shared__ __align__(16) bf16_t As[2][TM][LDA];
    __shared__ __align__(16) bf16_t Bs[2][TN][LDA];

    const int tid  = threadIdx.x;
    const int wid  = tid >> 5;
    const int lane = tid & 31;
    const int bM = blockIdx.y * TM;
    const int bN = blockIdx.x * TN;
    const int waveM = (wid >> 2) * 64;   // 0 or 64
    const int waveN = (wid & 3) * 32;    // 0,32,64,96

    v8f acc[4][2];
#pragma unroll
    for (int mt = 0; mt < 4; ++mt)
#pragma unroll
        for (int nt = 0; nt < 2; ++nt)
#pragma unroll
            for (int r = 0; r < 8; ++r) acc[mt][nt][r] = 0.f;

    const int lm = lane & 15;           // row within 16-wide fragment slice
    const int kb = (lane >> 4) * 8;     // k base (0 or 8; +16 for hi chunk)

    // each thread owns two 16B chunks per tile: rows r0 and r0+64, k-offset k0
    const int r0 = tid >> 2;
    const int k0 = (tid & 3) * 8;

    auto stage = [&](int kc, int b) {
        async_copy_b128(&As[b][r0][k0],      &xb [(size_t)(bM + r0)      * KP + kc + k0]);
        async_copy_b128(&As[b][r0 + 64][k0], &xb [(size_t)(bM + r0 + 64) * KP + kc + k0]);
        async_copy_b128(&Bs[b][r0][k0],      &wb1[(size_t)(bN + r0)      * KP + kc + k0]);
        async_copy_b128(&Bs[b][r0 + 64][k0], &wb1[(size_t)(bN + r0 + 64) * KP + kc + k0]);
    };

    stage(0, 0);
    wait_async0();
    __syncthreads();

    for (int kc = 0, it = 0; kc < KP; kc += TK, ++it) {
        const int cur = it & 1;
        if (kc + TK < KP) stage(kc + TK, cur ^ 1);   // DMA next tile while computing

        v16bf af[4], bfr[2];
#pragma unroll
        for (int mt = 0; mt < 4; ++mt) {
            const bf16_t* src = &As[cur][waveM + mt * 16 + lm][kb];
#pragma unroll
            for (int e = 0; e < 8; ++e) { af[mt][e] = src[e]; af[mt][8 + e] = src[16 + e]; }
        }
#pragma unroll
        for (int nt = 0; nt < 2; ++nt) {
            const bf16_t* src = &Bs[cur][waveN + nt * 16 + lm][kb];
#pragma unroll
            for (int e = 0; e < 8; ++e) { bfr[nt][e] = src[e]; bfr[nt][8 + e] = src[16 + e]; }
        }
#pragma unroll
        for (int mt = 0; mt < 4; ++mt)
#pragma unroll
            for (int nt = 0; nt < 2; ++nt)
                acc[mt][nt] = __builtin_amdgcn_wmma_f32_16x16x32_bf16(
                    false, af[mt], false, bfr[nt], (short)0, acc[mt][nt], false, false);

        wait_async0();       // next tile's DMA landed in the idle buffer
        __syncthreads();     // all waves done reading current buffer
    }

    // epilogue: scale by alpha1[n], store h (C layout: lane&15 = N,
    // VGPR r -> M = r + 8*(lane>>4))
    const int mhalf = (lane >> 4) * 8;
#pragma unroll
    for (int nt = 0; nt < 2; ++nt) {
        const int gn = bN + waveN + nt * 16 + lm;
        const float a = alpha1[gn];
#pragma unroll
        for (int mt = 0; mt < 4; ++mt) {
#pragma unroll
            for (int r = 0; r < 8; ++r) {
                const int gm = bM + waveM + mt * 16 + mhalf + r;
                h[(size_t)gm * HID + gn] = a * acc[mt][nt][r];
            }
        }
    }
}

// ---------------- per-feature sum / sumsq over the batch -------------------
__global__ __launch_bounds__(256)
void k_colreduce(const float* __restrict__ h, float* __restrict__ sum,
                 float* __restrict__ sumsq) {
    const int f = blockIdx.x * 256 + threadIdx.x;   // gridDim.x = 4
    const int r0 = blockIdx.y * 256;                // gridDim.y = 256
    float s = 0.f, s2 = 0.f;
    for (int r = r0; r < r0 + 256; ++r) {
        float v = h[(size_t)r * HID + f];
        s += v; s2 += v * v;
    }
    atomicAdd(&sum[f], s);
    atomicAdd(&sumsq[f], s2);
}

// ---------------- BN finalize: a = sign(bnA*h + bnB) -----------------------
__global__ __launch_bounds__(256)
void k_bnfinal(const float* __restrict__ sum, const float* __restrict__ sumsq,
               const float* __restrict__ gamma, const float* __restrict__ beta,
               float* __restrict__ bnA, float* __restrict__ bnB) {
    const int f = blockIdx.x * 256 + threadIdx.x;
    const float mu  = sum[f]   / (float)Bsz;
    const float var = sumsq[f] / (float)Bsz - mu * mu;
    const float rstd = rsqrtf(var + BN_EPS);
    const float A = gamma[f] * rstd;
    bnA[f] = A;
    bnB[f] = beta[f] - A * mu;
}

// ---------------- binarize + XNOR-popcount GEMM2 ---------------------------
__global__ __launch_bounds__(256)
void k_binact_gemm2(const float* __restrict__ h, const float* __restrict__ bnA,
                    const float* __restrict__ bnB,
                    const unsigned* __restrict__ w2bits,
                    const float* __restrict__ alpha2, float* __restrict__ out) {
    __shared__ unsigned wbits_s[NCLS * 32];
    __shared__ float a2_s[NCLS];
    __shared__ float bnA_s[HID];
    __shared__ float bnB_s[HID];
    for (int i = threadIdx.x; i < NCLS * 32; i += 256) wbits_s[i] = w2bits[i];
    if (threadIdx.x < NCLS) a2_s[threadIdx.x] = alpha2[threadIdx.x];
    for (int i = threadIdx.x; i < HID; i += 256) { bnA_s[i] = bnA[i]; bnB_s[i] = bnB[i]; }
    __syncthreads();

    const int wid  = threadIdx.x >> 5;
    const int lane = threadIdx.x & 31;
    const int b = blockIdx.x * 8 + wid;

    unsigned myword = 0u;
#pragma unroll
    for (int i = 0; i < 32; ++i) {
        const int f = i * 32 + lane;
        const float v = h[(size_t)b * HID + f];
        const bool bit = (bnA_s[f] * v + bnB_s[f] >= 0.f);
        const unsigned m = __builtin_amdgcn_ballot_w32(bit);
        if (lane == i) myword = m;
    }
#pragma unroll
    for (int c = 0; c < NCLS; ++c) {
        int cnt = __popc(myword ^ wbits_s[c * 32 + lane]);
#pragma unroll
        for (int off = 16; off > 0; off >>= 1) cnt += __shfl_xor(cnt, off, 32);
        if (lane == 0) out[(size_t)b * NCLS + c] = a2_s[c] * (float)(HID - 2 * cnt);
    }
}

extern "C" void kernel_launch(void* const* d_in, const int* in_sizes, int n_in,
                              void* d_out, int out_size, void* d_ws, size_t ws_size,
                              hipStream_t stream) {
    const float* x     = (const float*)d_in[0];   // [65536, 784]
    const float* w1    = (const float*)d_in[1];   // [1024, 784]
    const float* w2    = (const float*)d_in[2];   // [10, 1024]
    const float* gamma = (const float*)d_in[3];   // [1024]
    const float* beta  = (const float*)d_in[4];   // [1024]
    float* out = (float*)d_out;                   // [65536, 10]

    // workspace layout (16B+ aligned chunks)
    char* ws = (char*)d_ws;
    size_t off = 0;
    bf16_t*   wb1    = (bf16_t*)(ws + off);  off += (size_t)HID * KP * sizeof(bf16_t); // 1.6 MB
    float*    alpha1 = (float*)(ws + off);   off += 4096;
    float*    alpha2 = (float*)(ws + off);   off += 256;
    unsigned* w2bits = (unsigned*)(ws + off);off += 2048;
    float*    sum    = (float*)(ws + off);   off += 4096;
    float*    sumsq  = (float*)(ws + off);   off += 4096;
    float*    bnA    = (float*)(ws + off);   off += 4096;
    float*    bnB    = (float*)(ws + off);   off += 4096;
    bf16_t*   xb     = (bf16_t*)(ws + off);  off += (size_t)Bsz * KP * sizeof(bf16_t); // 105 MB
    float*    hbuf   = (float*)(ws + off);   off += (size_t)Bsz * HID * sizeof(float); // 256 MB

    k_init_stats<<<1, 256, 0, stream>>>(sum, sumsq);
    k_convert_x<<<(Bsz * (KP / 8)) / 256, 256, 0, stream>>>(x, xb);
    k_prep_w1<<<HID, 256, 0, stream>>>(w1, wb1, alpha1);
    k_prep_w2<<<NCLS, 256, 0, stream>>>(w2, w2bits, alpha2);
    k_gemm1_wmma<<<dim3(HID / TN, Bsz / TM), 256, 0, stream>>>(xb, wb1, alpha1, hbuf);
    k_colreduce<<<dim3(HID / 256, Bsz / 256), 256, 0, stream>>>(hbuf, sum, sumsq);
    k_bnfinal<<<HID / 256, 256, 0, stream>>>(sum, sumsq, gamma, beta, bnA, bnB);
    k_binact_gemm2<<<Bsz / 8, 256, 0, stream>>>(hbuf, bnA, bnB, w2bits, alpha2, out);
}
